// GroupedQueryAttention_77326591197230
// MI455X (gfx1250) — compile-verified, matrix operand order checked by reading, not run on hardware
//
#include <hip/hip_runtime.h>
#include <math.h>

// ---------------------------------------------------------------------------
// Types for CDNA5 WMMA (wave32).  All LDS access goes through explicit
// addrspace(3) pointers so the backend always emits ds_* (never flat_*).
// ---------------------------------------------------------------------------
#define AS3 __attribute__((address_space(3)))
typedef __bf16 bf16x4 __attribute__((ext_vector_type(4)));
typedef __bf16 bf16x8 __attribute__((ext_vector_type(8)));
typedef __bf16 v16bf  __attribute__((ext_vector_type(16)));
typedef float  v8f    __attribute__((ext_vector_type(8)));
typedef AS3 __bf16  lbf16;
typedef AS3 bf16x4  lbf16x4_t;
typedef AS3 bf16x8  lbf16x8_t;

#define B_SZ   2
#define S_SZ   2048
#define H_SZ   4096
#define NH     32
#define NKV    8
#define HD     128
#define SCALE  0.0883883476483184f   // 1/sqrt(128)
#define LN_ROPE_BASE 13.1223633774f  // ln(500000)

#define LDS_BASE 64u   // keep all LDS offsets nonzero (AS(3) null != 0) & aligned

// A-fragment load from LDS: lane holds row (lane&15), K elements
// {kb..kb+7, kb+16..kb+23}, kb = (lane>>4)*8.  base = &lds[row*stride + k0 + kb]
__device__ __forceinline__ v16bf load_frag(const lbf16* base) {
    bf16x8 lo = *(const lbf16x8_t*)(base);
    bf16x8 hi = *(const lbf16x8_t*)(base + 16);
    return __builtin_shufflevector(lo, hi, 0, 1, 2, 3, 4, 5, 6, 7,
                                           8, 9, 10, 11, 12, 13, 14, 15);
}

__device__ __forceinline__ v8f wmma_bf16(v16bf a, v16bf b, v8f c) {
    return __builtin_amdgcn_wmma_f32_16x16x32_bf16(
        /*neg_a=*/false, a, /*neg_b=*/false, b,
        /*c_mod=*/(short)0, c, /*reuse_a=*/false, /*reuse_b=*/false);
}

// CDNA5 async global->LDS copy (16B per active lane), ASYNCcnt-tracked.
__device__ __forceinline__ void async_copy_b128(unsigned int lds_off,
                                                unsigned long long gaddr) {
    asm volatile("global_load_async_to_lds_b128 %0, %1, off"
                 :: "v"(lds_off), "v"(gaddr) : "memory");
}
__device__ __forceinline__ void wait_async0() {
    asm volatile("s_wait_asynccnt 0" ::: "memory");
}

// ---------------------------------------------------------------------------
// GEMM tiling common (128x128 tile, K-chunk 64, 8 waves)
// ---------------------------------------------------------------------------
#define GK   64      // K-chunk per LDS stage
#define GST  72      // LDS row stride (elems): 64 + 8 pad -> conflict-free b128
#define GTILE (128 * GST)
// LDS layout (byte offsets): [A0][B0][A1][B1] after LDS_BASE
#define GA0 (LDS_BASE)
#define GB0 (LDS_BASE + 1u * GTILE * 2u)
#define GA1 (LDS_BASE + 2u * GTILE * 2u)
#define GB1 (LDS_BASE + 3u * GTILE * 2u)
#define GEMM_SMEM (LDS_BASE + 4u * GTILE * 2u)

// stage a 128x64 f32 tile -> bf16 LDS (each thread: half a row = 32 elems)
__device__ __forceinline__ void stage_f32(const float* __restrict__ src_row,
                                          lbf16* dst, int rowL, int colL) {
    const float4* src = (const float4*)src_row;
    lbf16* d = dst + rowL * GST + colL;
    #pragma unroll
    for (int i = 0; i < 8; ++i) {
        float4 f = src[i];
        bf16x4 h = { (__bf16)f.x, (__bf16)f.y, (__bf16)f.z, (__bf16)f.w };
        *(lbf16x4_t*)(d + i * 4) = h;
    }
}

// 16 WMMAs on one staged (A,B) tile pair.  B-fragments are preloaded in a
// batch so a single s_wait_dscnt covers 8 back-to-back WMMAs.
__device__ __forceinline__ void mma_tile(const lbf16* As, const lbf16* Bs,
                                         v8f* acc, int wid, int lo16, int kb) {
    #pragma unroll
    for (int kk = 0; kk < 2; ++kk) {
        v16bf a = load_frag(As + (wid * 16 + lo16) * GST + kk * 32 + kb);
        v16bf bfr[8];
        #pragma unroll
        for (int t = 0; t < 8; ++t)
            bfr[t] = load_frag(Bs + (t * 16 + lo16) * GST + kk * 32 + kb);
        #pragma unroll
        for (int t = 0; t < 8; ++t)
            acc[t] = wmma_bf16(a, bfr[t], acc[t]);
    }
}

// ---------------------------------------------------------------------------
// Kernel 1: QKV projection  out = X @ W^T  (f32 in, bf16 head-major out)
// Double-buffered dynamic LDS; loop unrolled x2 so every LDS address is a
// compile-time-constant offset (pure ds_load/ds_store codegen).
// ---------------------------------------------------------------------------
__global__ __launch_bounds__(256) void qkv_gemm_kernel(
    const float* __restrict__ X, const float* __restrict__ W,
    __bf16* __restrict__ out, int nheads, int apply_rope)
{
    lbf16* A0 = (lbf16*)(GA0);
    lbf16* B0 = (lbf16*)(GB0);
    lbf16* A1 = (lbf16*)(GA1);
    lbf16* B1 = (lbf16*)(GB1);

    const int tid  = threadIdx.x;
    const int lane = tid & 31;
    const int wid  = tid >> 5;
    const int lo16 = lane & 15;
    const int hi16 = lane >> 4;
    const int kb   = hi16 * 8;
    const int m0   = blockIdx.y * 128;
    const int n0   = blockIdx.x * 128;

    const int rowL = tid >> 1;           // 0..127
    const int colL = (tid & 1) * 32;     // 0 / 32

    const float* Arow = X + (size_t)(m0 + rowL) * H_SZ + colL;
    const float* Brow = W + (size_t)(n0 + rowL) * H_SZ + colL;

    v8f acc[8] = {};

    stage_f32(Arow, A0, rowL, colL);
    stage_f32(Brow, B0, rowL, colL);
    __syncthreads();

    const int NSTEP = H_SZ / GK;         // 64 (even)
    for (int step = 0; step < NSTEP; step += 2) {
        // compute buf0, stage step+1 into buf1
        if (step + 1 < NSTEP) {
            stage_f32(Arow + (step + 1) * GK, A1, rowL, colL);
            stage_f32(Brow + (step + 1) * GK, B1, rowL, colL);
        }
        mma_tile(A0, B0, acc, wid, lo16, kb);
        __syncthreads();
        // compute buf1, stage step+2 into buf0
        if (step + 2 < NSTEP) {
            stage_f32(Arow + (step + 2) * GK, A0, rowL, colL);
            stage_f32(Brow + (step + 2) * GK, B0, rowL, colL);
        }
        mma_tile(A1, B1, acc, wid, lo16, kb);
        __syncthreads();
    }

    // ---- epilogue: optional RoPE (cols d and d+64 are acc[t] / acc[t+4]) ----
    if (apply_rope) {
        #pragma unroll
        for (int t = 0; t < 4; ++t) {
            int d = t * 16 + lo16;                       // 0..63
            float inv_freq = __expf(-(float)d * (LN_ROPE_BASE / 64.0f));
            #pragma unroll
            for (int v = 0; v < 8; ++v) {
                int row = m0 + wid * 16 + v + 8 * hi16;
                int s   = row & (S_SZ - 1);
                float ang = (float)s * inv_freq;
                float c = __cosf(ang), sn = __sinf(ang);
                float x0 = acc[t][v], x1 = acc[t + 4][v];
                acc[t][v]     = x0 * c - x1 * sn;
                acc[t + 4][v] = x1 * c + x0 * sn;
            }
        }
    }

    // ---- store head-major [B, nheads, S, HD] bf16 ----
    const int h = n0 >> 7;   // HD = 128, tile spans exactly one head
    #pragma unroll
    for (int t = 0; t < 8; ++t) {
        int dcol = t * 16 + lo16;
        #pragma unroll
        for (int v = 0; v < 8; ++v) {
            int row  = m0 + wid * 16 + v + 8 * hi16;
            int s    = row & (S_SZ - 1);
            int bidx = row >> 11;
            size_t idx = (((size_t)bidx * nheads + h) * S_SZ + s) * HD + dcol;
            out[idx] = (__bf16)acc[t][v];
        }
    }
}

// ---------------------------------------------------------------------------
// Kernel 2: flash attention (non-causal, GQA kvh = qh/4)
// grid = B*NH*(S/128) = 1024 blocks, block = 256 (8 waves)
// LDS: Ks | Vt | Ps | Qs, each 128x136 bf16 (constant byte offsets).
// Q/K tiles staged with CDNA5 async global->LDS copies.
// ---------------------------------------------------------------------------
#define AST     136
#define ATSZ    (128 * AST)                         // elems per tile
#define KS_OFF  (LDS_BASE)
#define VT_OFF  (LDS_BASE + 1u * ATSZ * 2u)
#define PS_OFF  (LDS_BASE + 2u * ATSZ * 2u)
#define QS_OFF  (LDS_BASE + 3u * ATSZ * 2u)
#define ATTN_SMEM (LDS_BASE + 4u * ATSZ * 2u)

__global__ __launch_bounds__(256) void attn_kernel(
    const __bf16* __restrict__ Q,   // [B,NH,S,HD]
    const __bf16* __restrict__ K,   // [B,NKV,S,HD]
    const __bf16* __restrict__ V,   // [B,NKV,S,HD]
    __bf16* __restrict__ O)         // [B,S,NH*HD]
{
    lbf16* Ks = (lbf16*)(KS_OFF);   // K tile  [kpos][d]
    lbf16* Vt = (lbf16*)(VT_OFF);   // V tile transposed [d][kpos]
    lbf16* Ps = (lbf16*)(PS_OFF);   // P tile  [qrow][kpos]
    lbf16* Qs = (lbf16*)(QS_OFF);   // Q stage [qrow][d]

    const int tid  = threadIdx.x;
    const int lane = tid & 31;
    const int wid  = tid >> 5;
    const int lo16 = lane & 15;
    const int hi16 = lane >> 4;
    const int kb   = hi16 * 8;

    const int blk = blockIdx.x;
    const int mt  = blk & 15;                 // query tile (S/128 = 16)
    const int qh  = (blk >> 4) & 31;
    const int b   = blk >> 9;
    const int kvh = qh >> 2;                  // GROUPS = 4
    const int m0  = mt * 128;

    const int rowL = tid >> 1;                // 0..127
    const int colL = (tid & 1) * 64;          // 0 / 64

    // ---- async-stage Q tile ----
    {
        unsigned int loff = QS_OFF + (unsigned)(rowL * AST + colL) * 2;
        unsigned long long ga = (unsigned long long)(uintptr_t)
            (Q + (((size_t)(b * NH + qh)) * S_SZ + m0 + rowL) * HD + colL);
        #pragma unroll
        for (int i = 0; i < 8; ++i) async_copy_b128(loff + i * 16, ga + i * 16);
    }
    wait_async0();
    __syncthreads();
    v16bf qa[4];
    #pragma unroll
    for (int kk = 0; kk < 4; ++kk)
        qa[kk] = load_frag(Qs + (wid * 16 + lo16) * AST + kk * 32 + kb);

    v8f   oacc[8] = {};
    float mrow[8], lrow[8];
    #pragma unroll
    for (int v = 0; v < 8; ++v) { mrow[v] = -3.0e38f; lrow[v] = 0.0f; }

    const __bf16* Kg = K + ((size_t)(b * NKV + kvh)) * S_SZ * HD;
    const __bf16* Vg = V + ((size_t)(b * NKV + kvh)) * S_SZ * HD;

    for (int kt = 0; kt < S_SZ / 128; ++kt) {
        __syncthreads();   // previous iteration done reading Ks/Vt/Ps
        // ---- async-stage K tile ----
        {
            unsigned int loff = KS_OFF + (unsigned)(rowL * AST + colL) * 2;
            unsigned long long ga = (unsigned long long)(uintptr_t)
                (Kg + (size_t)(kt * 128 + rowL) * HD + colL);
            #pragma unroll
            for (int i = 0; i < 8; ++i) async_copy_b128(loff + i * 16, ga + i * 16);
        }
        // ---- load V tile transposed into Vt[d][kpos] (VGPR path) ----
        {
            const bf16x8* src = (const bf16x8*)(Vg + (size_t)(kt * 128 + rowL) * HD + colL);
            #pragma unroll
            for (int i = 0; i < 8; ++i) {
                bf16x8 x = src[i];
                #pragma unroll
                for (int j = 0; j < 8; ++j)
                    Vt[(colL + i * 8 + j) * AST + rowL] = x[j];
            }
        }
        // prefetch next K/V tiles into cache while we compute
        if (kt + 1 < S_SZ / 128) {
            __builtin_prefetch(Kg + (size_t)((kt + 1) * 128 + rowL) * HD + colL, 0, 3);
            __builtin_prefetch(Vg + (size_t)((kt + 1) * 128 + rowL) * HD + colL, 0, 3);
        }
        wait_async0();
        __syncthreads();

        // ---- S = Q @ K^T  (B-frag = rows of K; preload 4-frag batches) ----
        v8f sacc[8] = {};
        #pragma unroll
        for (int kk = 0; kk < 4; ++kk) {
            #pragma unroll
            for (int half = 0; half < 2; ++half) {
                v16bf kf[4];
                #pragma unroll
                for (int t4 = 0; t4 < 4; ++t4)
                    kf[t4] = load_frag(Ks + ((half * 4 + t4) * 16 + lo16) * AST + kk * 32 + kb);
                #pragma unroll
                for (int t4 = 0; t4 < 4; ++t4)
                    sacc[half * 4 + t4] = wmma_bf16(qa[kk], kf[t4], sacc[half * 4 + t4]);
            }
        }

        // ---- online softmax (rows striped: row = v + 8*hi16, col = lo16) ----
        #pragma unroll
        for (int v = 0; v < 8; ++v) {
            float tm = -3.0e38f;
            #pragma unroll
            for (int t = 0; t < 8; ++t) tm = fmaxf(tm, sacc[t][v]);
            #pragma unroll
            for (int msk = 1; msk <= 8; msk <<= 1)
                tm = fmaxf(tm, __shfl_xor(tm, msk, 32));
            tm *= SCALE;
            float mnew = fmaxf(mrow[v], tm);
            float corr = __expf(mrow[v] - mnew);
            float rs = 0.0f;
            #pragma unroll
            for (int t = 0; t < 8; ++t) {
                float p = __expf(sacc[t][v] * SCALE - mnew);
                sacc[t][v] = p;
                rs += p;
            }
            #pragma unroll
            for (int msk = 1; msk <= 8; msk <<= 1)
                rs += __shfl_xor(rs, msk, 32);
            lrow[v] = lrow[v] * corr + rs;
            mrow[v] = mnew;
            #pragma unroll
            for (int t = 0; t < 8; ++t) oacc[t][v] *= corr;
        }

        // ---- write P (C-layout -> LDS, wave-local rows; LDS ops in-order) ----
        #pragma unroll
        for (int t = 0; t < 8; ++t)
            #pragma unroll
            for (int v = 0; v < 8; ++v)
                Ps[(wid * 16 + v + 8 * hi16) * AST + t * 16 + lo16] = (__bf16)sacc[t][v];

        // ---- O += P @ V  (A-frag from Ps, B-frag = rows of Vt = cols of V) ----
        #pragma unroll
        for (int kk = 0; kk < 4; ++kk) {
            v16bf pa = load_frag(Ps + (wid * 16 + lo16) * AST + kk * 32 + kb);
            #pragma unroll
            for (int half = 0; half < 2; ++half) {
                v16bf vf[4];
                #pragma unroll
                for (int t4 = 0; t4 < 4; ++t4)
                    vf[t4] = load_frag(Vt + ((half * 4 + t4) * 16 + lo16) * AST + kk * 32 + kb);
                #pragma unroll
                for (int t4 = 0; t4 < 4; ++t4)
                    oacc[half * 4 + t4] = wmma_bf16(pa, vf[t4], oacc[half * 4 + t4]);
            }
        }
    }

    // ---- normalize and store [B, S, NH*HD] bf16 ----
    #pragma unroll
    for (int v = 0; v < 8; ++v) lrow[v] = 1.0f / lrow[v];
    #pragma unroll
    for (int t = 0; t < 8; ++t) {
        int col = qh * HD + t * 16 + lo16;
        #pragma unroll
        for (int v = 0; v < 8; ++v) {
            int s = m0 + wid * 16 + v + 8 * hi16;
            size_t idx = ((size_t)b * S_SZ + s) * (NH * HD) + col;
            O[idx] = (__bf16)(oacc[t][v] * lrow[v]);
        }
    }
}

// ---------------------------------------------------------------------------
// Kernel 3: output projection  out = A @ Wo^T  (A bf16, Wo f32, out f32)
// Double-buffered; A tiles staged with async global->LDS (already bf16).
// ---------------------------------------------------------------------------
__global__ __launch_bounds__(256) void oproj_gemm_kernel(
    const __bf16* __restrict__ A, const float* __restrict__ W,
    float* __restrict__ out)
{
    lbf16* A0 = (lbf16*)(GA0);
    lbf16* B0 = (lbf16*)(GB0);
    lbf16* A1 = (lbf16*)(GA1);
    lbf16* B1 = (lbf16*)(GB1);

    const int tid  = threadIdx.x;
    const int lane = tid & 31;
    const int wid  = tid >> 5;
    const int lo16 = lane & 15;
    const int hi16 = lane >> 4;
    const int kb   = hi16 * 8;
    const int m0   = blockIdx.y * 128;
    const int n0   = blockIdx.x * 128;

    const int rowL = tid >> 1;
    const int colL = (tid & 1) * 32;

    const __bf16* Arow = A + (size_t)(m0 + rowL) * H_SZ + colL;
    const float*  Brow = W + (size_t)(n0 + rowL) * H_SZ + colL;
    const unsigned int lds_row = (unsigned)(rowL * GST + colL) * 2;

    v8f acc[8] = {};

    // prologue stage (buffer 0): A async, B convert
    {
        unsigned long long ga = (unsigned long long)(uintptr_t)Arow;
        #pragma unroll
        for (int i = 0; i < 4; ++i) async_copy_b128(GA0 + lds_row + i * 16, ga + i * 16);
    }
    stage_f32(Brow, B0, rowL, colL);
    wait_async0();
    __syncthreads();

    const int NSTEP = H_SZ / GK;   // 64 (even)
    for (int step = 0; step < NSTEP; step += 2) {
        if (step + 1 < NSTEP) {
            unsigned long long ga = (unsigned long long)(uintptr_t)(Arow + (step + 1) * GK);
            #pragma unroll
            for (int i = 0; i < 4; ++i) async_copy_b128(GA1 + lds_row + i * 16, ga + i * 16);
            stage_f32(Brow + (step + 1) * GK, B1, rowL, colL);
        }
        mma_tile(A0, B0, acc, wid, lo16, kb);
        wait_async0();
        __syncthreads();

        if (step + 2 < NSTEP) {
            unsigned long long ga = (unsigned long long)(uintptr_t)(Arow + (step + 2) * GK);
            #pragma unroll
            for (int i = 0; i < 4; ++i) async_copy_b128(GA0 + lds_row + i * 16, ga + i * 16);
            stage_f32(Brow + (step + 2) * GK, B0, rowL, colL);
        }
        mma_tile(A1, B1, acc, wid, lo16, kb);
        wait_async0();
        __syncthreads();
    }

    #pragma unroll
    for (int t = 0; t < 8; ++t) {
        int col = n0 + t * 16 + lo16;
        #pragma unroll
        for (int v = 0; v < 8; ++v) {
            int row = m0 + wid * 16 + v + 8 * hi16;
            out[(size_t)row * H_SZ + col] = acc[t][v];
        }
    }
}

// ---------------------------------------------------------------------------
// Host launcher
// ---------------------------------------------------------------------------
extern "C" void kernel_launch(void* const* d_in, const int* in_sizes, int n_in,
                              void* d_out, int out_size, void* d_ws, size_t ws_size,
                              hipStream_t stream) {
    (void)in_sizes; (void)n_in; (void)out_size; (void)ws_size;

    const float* X  = (const float*)d_in[0];   // [B,S,H]
    const float* Wq = (const float*)d_in[1];   // [NH*HD, H]
    const float* Wk = (const float*)d_in[2];   // [NKV*HD, H]
    const float* Wv = (const float*)d_in[3];   // [NKV*HD, H]
    const float* Wo = (const float*)d_in[4];   // [H, NH*HD]
    float* out = (float*)d_out;                // [B,S,H] f32

    // workspace carve (bf16): Q | K | V | attn_out  (~84 MB total)
    __bf16* Qb = (__bf16*)d_ws;
    __bf16* Kb = Qb + (size_t)B_SZ * NH  * S_SZ * HD;
    __bf16* Vb = Kb + (size_t)B_SZ * NKV * S_SZ * HD;
    __bf16* Ab = Vb + (size_t)B_SZ * NKV * S_SZ * HD;

    const dim3 blk(256);

    // QKV projections (RoPE fused for Q and K)
    qkv_gemm_kernel<<<dim3(NH * HD / 128, B_SZ * S_SZ / 128), blk, GEMM_SMEM, stream>>>(
        X, Wq, Qb, NH, 1);
    qkv_gemm_kernel<<<dim3(NKV * HD / 128, B_SZ * S_SZ / 128), blk, GEMM_SMEM, stream>>>(
        X, Wk, Kb, NKV, 1);
    qkv_gemm_kernel<<<dim3(NKV * HD / 128, B_SZ * S_SZ / 128), blk, GEMM_SMEM, stream>>>(
        X, Wv, Vb, NKV, 0);

    // flash attention: 1024 blocks
    attn_kernel<<<dim3(B_SZ * NH * (S_SZ / 128)), blk, ATTN_SMEM, stream>>>(Qb, Kb, Vb, Ab);

    // output projection -> f32 result
    oproj_gemm_kernel<<<dim3(H_SZ / 128, B_SZ * S_SZ / 128), blk, GEMM_SMEM, stream>>>(Ab, Wo, out);
}
